// BasicBlock_1w1a_23570780521118
// MI455X (gfx1250) — matrix operand order repaired and checked by reading, not compiled, not run on hardware
//
#include <hip/hip_runtime.h>
#include <hip/hip_bf16.h>

// ---------------------------------------------------------------------------
// AdaBin BasicBlock (1w1a) on gfx1250:
//   conv -> bn -> +res -> maxout  (x2), conv lowered to implicit GEMM on
//   v_wmma_f32_16x16x32_f16.  M=64*28*28=50176, N=256, K=256*9=2304.
//   Each wave owns a 16x32 C tile (2 accumulators) and sets the CDNA5
//   reuse_a VOP3P hint on the second WMMA sharing the A fragment.
// ---------------------------------------------------------------------------

typedef _Float16 v8h  __attribute__((ext_vector_type(8)));
typedef _Float16 v16h __attribute__((ext_vector_type(16)));
typedef float    v8f  __attribute__((ext_vector_type(8)));

#define BATCH   64
#define CH      256
#define HH      28
#define WW      28
#define HP      30      // padded
#define WP      30
#define PIX     (HH*WW)                 // 784
#define MROWS   (BATCH*PIX)             // 50176
#define KTOT    (CH*9)                  // 2304
#define NGRPS   (CH/16)                 // 16
#define PAD_ELEMS ((size_t)BATCH*HP*WP*CH)   // 14,745,600 f16

// ---------------------------------------------------------------------------
// Zero the padded activation buffer (vectorized b128 stores).
// ---------------------------------------------------------------------------
__global__ void zero_pad_kernel(_Float16* __restrict__ p, long nvec) {
    long i = (long)blockIdx.x * blockDim.x + threadIdx.x;
    if (i < nvec) {
        v8h z = {};
        ((v8h*)p)[i] = z;
    }
}

// ---------------------------------------------------------------------------
// Activation binarization: NCHW f32 -> padded NHWC f16.
//   xa = sign((x - beta)/alpha) * alpha + beta   (sign(0)=0, matching jnp)
// Reads are fully coalesced (NCHW-linear); 2B scattered writes land in L2.
// ---------------------------------------------------------------------------
__global__ void binarize_act_kernel(const float* __restrict__ x,
                                    const float* __restrict__ alpha,
                                    const float* __restrict__ beta,
                                    _Float16* __restrict__ xa_pad, int total) {
    int idx = blockIdx.x * blockDim.x + threadIdx.x;
    if (idx >= total) return;
    int ww = idx % WW;
    int t  = idx / WW;
    int hh = t % HH;  t /= HH;
    int c  = t % CH;
    int n  = t / CH;
    float a  = alpha[c];
    float bt = beta[c];
    float r  = (x[idx] - bt) / a;
    float s  = (r > 0.f) ? 1.f : ((r < 0.f) ? -1.f : 0.f);
    float v  = s * a + bt;
    size_t o = (((size_t)n * (HP*WP)) + (size_t)(hh + 1) * WP + (ww + 1)) * CH + c;
    xa_pad[o] = (_Float16)v;
}

// ---------------------------------------------------------------------------
// Weight binarization + packing into per-lane WMMA B-fragment layout.
// One block per output filter o.  K is reordered as k = (kh*3+kw)*256 + c.
// bpk[((t*16 + o/16)*32 + lane)*16 + i] = wb[K = t*32 + (lane/16)*16 + i][o]
// so each GEMM lane reads one contiguous 32B fragment per K-chunk.
// ---------------------------------------------------------------------------
__global__ __launch_bounds__(256)
void pack_weights_kernel(const float* __restrict__ w, _Float16* __restrict__ bpk) {
    int o   = blockIdx.x;
    int tid = threadIdx.x;
    const float* wo = w + (size_t)o * KTOT;

    float s = 0.f, s2 = 0.f;
    for (int k = tid; k < KTOT; k += 256) {
        float v = wo[k];
        s += v; s2 += v * v;
    }
    __shared__ float sh[256];
    __shared__ float sh2[256];
    sh[tid] = s; sh2[tid] = s2;
    __syncthreads();
    for (int st = 128; st > 0; st >>= 1) {
        if (tid < st) { sh[tid] += sh[tid + st]; sh2[tid] += sh2[tid + st]; }
        __syncthreads();
    }
    float betaW  = sh[0] / (float)KTOT;
    float varW   = sh2[0] / (float)KTOT - betaW * betaW;
    float alphaW = sqrtf(fmaxf(varW, 0.f));

    for (int k = tid; k < KTOT; k += 256) {
        int c  = k & 255;       // channel
        int g  = k >> 8;        // kh*3+kw
        float wv  = wo[c * 9 + g];          // OIHW source
        float sgn = (wv > betaW) ? 1.f : ((wv < betaW) ? -1.f : 0.f);
        float wb  = sgn * alphaW + betaW;
        int t    = k >> 5;
        int krel = k & 31;
        int lane = ((krel >> 4) << 4) | (o & 15);
        int i    = krel & 15;
        size_t idx = ((((size_t)t * NGRPS) + (o >> 4)) * 32 + lane) * 16 + i;
        bpk[idx] = (_Float16)wb;
    }
}

// ---------------------------------------------------------------------------
// Implicit-GEMM conv on v_wmma_f32_16x16x32_f16 with fused
// BN-eval + residual-add + maxout epilogue.
// Block = 256 threads = 8 waves; wave (mw,nw) owns a 16x32 C tile:
//   rows  [blockIdx.x*32 + mw*16, +16)
//   cols  [ (blockIdx.y*8 + nw*2)*16, +32 )   (two 16-wide n-groups)
// The A fragment is loaded once per K-chunk and fed to two WMMAs; the
// second sets reuse_a (identical instruction, same A operand).
// ---------------------------------------------------------------------------
__global__ __launch_bounds__(256)
void adabin_wmma_gemm_kernel(const _Float16* __restrict__ xa,   // [64][30][30][256]
                             const _Float16* __restrict__ bpk,  // packed B fragments
                             const float* __restrict__ res,     // residual, NCHW f32
                             const float* __restrict__ bn_g,
                             const float* __restrict__ bn_b,
                             const float* __restrict__ bn_m,
                             const float* __restrict__ bn_v,
                             const float* __restrict__ pos,
                             const float* __restrict__ neg,
                             float* __restrict__ out) {
    const int lane = threadIdx.x & 31;
    const int wv   = threadIdx.x >> 5;   // 0..7
    const int mw   = wv >> 2;            // 0..1
    const int nw   = wv & 3;             // 0..3

    const int mbase = blockIdx.x * 32 + mw * 16;
    const int jgrp0 = blockIdx.y * 8 + nw * 2;  // 0..15 over grid.y in {0,1}
    const int nbase = jgrp0 * 16;

    // A-row this lane feeds (ISA: A lanes 0-15 and 16-31 both cover M=0..15).
    const int mrow = mbase + (lane & 15);
    const int img  = mrow / PIX;
    const int pix  = mrow % PIX;
    const int ph   = pix / WW;           // output row -> padded row base
    const int pw   = pix % WW;
    const _Float16* abase =
        xa + (((size_t)img * (HP*WP)) + (size_t)ph * WP + pw) * CH;
    const int khalf = (lane >> 4) * 16;  // K sub-offset within 32-chunk

    v8f acc0 = {0.f, 0.f, 0.f, 0.f, 0.f, 0.f, 0.f, 0.f};
    v8f acc1 = {0.f, 0.f, 0.f, 0.f, 0.f, 0.f, 0.f, 0.f};

    union Frag { v16h v; v8h h[2]; };

    for (int g = 0; g < 9; ++g) {
        const _Float16* arow = abase + (size_t)((g / 3) * WP + (g % 3)) * CH;
        // pull the next 3x3-tap row toward the WGP cache (global_prefetch_b8)
        __builtin_prefetch(arow + CH, 0, 3);
#pragma unroll
        for (int tt = 0; tt < 8; ++tt) {
            const int c0 = tt * 32;
            Frag A, B0, B1;
            A.h[0] = *(const v8h*)(arow + c0 + khalf);
            A.h[1] = *(const v8h*)(arow + c0 + khalf + 8);
            const int t = g * 8 + tt;
            const _Float16* bp =
                bpk + ((((size_t)t * NGRPS) + jgrp0) * 32 + lane) * 16;
            B0.h[0] = *(const v8h*)(bp);
            B0.h[1] = *(const v8h*)(bp + 8);
            B1.h[0] = *(const v8h*)(bp + 32 * 16);       // jgrp0+1 fragment
            B1.h[1] = *(const v8h*)(bp + 32 * 16 + 8);
            acc0 = __builtin_amdgcn_wmma_f32_16x16x32_f16(
                false, A.v, false, B0.v, (short)0, acc0, false, false);
            acc1 = __builtin_amdgcn_wmma_f32_16x16x32_f16(
                false, A.v, false, B1.v, (short)0, acc1, /*reuse_a=*/true, false);
        }
    }

    // ---- epilogue: BN eval + residual + maxout -------------------------
    // C layout: lane L -> N = L%16 ; VGPR r -> M = r + 8*(L/16)
    const int o0 = nbase + (lane & 15);
    const int o1 = o0 + 16;
    const float inv0 = bn_g[o0] * rsqrtf(bn_v[o0] + 1e-5f);
    const float bia0 = bn_b[o0] - bn_m[o0] * inv0;
    const float p0   = pos[o0];
    const float q0   = neg[o0];
    const float inv1 = bn_g[o1] * rsqrtf(bn_v[o1] + 1e-5f);
    const float bia1 = bn_b[o1] - bn_m[o1] * inv1;
    const float p1   = pos[o1];
    const float q1   = neg[o1];
    const int mrowE = mbase + 8 * (lane >> 4);
#pragma unroll
    for (int r = 0; r < 8; ++r) {
        const int mg = mrowE + r;
        const int im = mg / PIX;
        const int px = mg % PIX;
        const size_t base = (size_t)im * CH * PIX + px;
        const size_t oi0 = base + (size_t)o0 * PIX;
        const size_t oi1 = base + (size_t)o1 * PIX;
        float v0 = acc0[r] * inv0 + bia0 + res[oi0];
        float v1 = acc1[r] * inv1 + bia1 + res[oi1];
        out[oi0] = p0 * fmaxf(v0, 0.f) - q0 * fmaxf(-v0, 0.f);
        out[oi1] = p1 * fmaxf(v1, 0.f) - q1 * fmaxf(-v1, 0.f);
    }
}

// ---------------------------------------------------------------------------
// Launcher
// ---------------------------------------------------------------------------
extern "C" void kernel_launch(void* const* d_in, const int* in_sizes, int n_in,
                              void* d_out, int out_size, void* d_ws, size_t ws_size,
                              hipStream_t stream) {
    const float* x       = (const float*)d_in[0];
    const float* conv1_w = (const float*)d_in[1];
    const float* alpha1  = (const float*)d_in[2];
    const float* beta1   = (const float*)d_in[3];
    const float* bn1_g   = (const float*)d_in[4];
    const float* bn1_b   = (const float*)d_in[5];
    const float* bn1_m   = (const float*)d_in[6];
    const float* bn1_v   = (const float*)d_in[7];
    const float* pos1    = (const float*)d_in[8];
    const float* neg1    = (const float*)d_in[9];
    const float* conv2_w = (const float*)d_in[10];
    const float* alpha2  = (const float*)d_in[11];
    const float* beta2   = (const float*)d_in[12];
    const float* bn2_g   = (const float*)d_in[13];
    const float* bn2_b   = (const float*)d_in[14];
    const float* bn2_m   = (const float*)d_in[15];
    const float* bn2_v   = (const float*)d_in[16];
    const float* pos2    = (const float*)d_in[17];
    const float* neg2    = (const float*)d_in[18];
    float* out = (float*)d_out;

    // workspace layout
    _Float16* xa_pad = (_Float16*)d_ws;                       // 29,491,200 B
    _Float16* bpk    = (_Float16*)((char*)d_ws + PAD_ELEMS * sizeof(_Float16));
                                                              // + 1,179,648 B

    const int total = MROWS * CH;          // 12,845,056 elements
    const long nvec = (long)(PAD_ELEMS / 8);

    dim3 gemm_grid(MROWS / 32, CH / 128);  // (1568, 2)

    // ---- layer 1 ----
    zero_pad_kernel<<<(int)((nvec + 255) / 256), 256, 0, stream>>>(xa_pad, nvec);
    binarize_act_kernel<<<(total + 255) / 256, 256, 0, stream>>>(
        x, alpha1, beta1, xa_pad, total);
    pack_weights_kernel<<<CH, 256, 0, stream>>>(conv1_w, bpk);
    adabin_wmma_gemm_kernel<<<gemm_grid, 256, 0, stream>>>(
        xa_pad, bpk, x, bn1_g, bn1_b, bn1_m, bn1_v, pos1, neg1, out);

    // ---- layer 2 (y1 lives in d_out: residual source + binarization input) ----
    binarize_act_kernel<<<(total + 255) / 256, 256, 0, stream>>>(
        out, alpha2, beta2, xa_pad, total);   // borders stay zero
    pack_weights_kernel<<<CH, 256, 0, stream>>>(conv2_w, bpk);
    adabin_wmma_gemm_kernel<<<gemm_grid, 256, 0, stream>>>(
        xa_pad, bpk, out, bn2_g, bn2_b, bn2_m, bn2_v, pos2, neg2, out);
}
